// MambaBlock_26645977104496
// MI455X (gfx1250) — matrix operand order repaired and checked by reading, not compiled, not run on hardware
//
#include <hip/hip_runtime.h>
#include <hip/hip_bf16.h>

// ---------------- types ----------------
typedef __bf16 bf16;
typedef __bf16 v16bf __attribute__((ext_vector_type(16)));
typedef float  v8f   __attribute__((ext_vector_type(8)));
typedef unsigned int u32x4 __attribute__((ext_vector_type(4)));

union FragAB { v16bf bf; u32x4 u[2]; };

// ---------------- problem constants ----------------
constexpr int BATCH   = 2;
constexpr int SEQ     = 1024;
constexpr int DMODEL  = 1024;
constexpr int DIN     = 2048;
constexpr int ROWS    = BATCH * SEQ;        // 2048
constexpr int DCONV   = 4;
constexpr int DTRANK  = 64;
constexpr int DSTATE  = 16;
constexpr int XDBLW   = DTRANK + 2 * DSTATE; // 96

// ---------------- workspace layout (bytes, all 256-aligned) ----------------
constexpr size_t OFF_WT_IN  = 0;                                           // bf16 [4096][1024]
constexpr size_t OFF_WT_X   = OFF_WT_IN  + (size_t)(2*DIN)*DMODEL*2;       // bf16 [96][2048]
constexpr size_t OFF_WT_DT  = OFF_WT_X   + (size_t)XDBLW*DIN*2;            // bf16 [2048][64]
constexpr size_t OFF_WT_OUT = OFF_WT_DT  + (size_t)DIN*DTRANK*2;           // bf16 [1024][2048]
constexpr size_t OFF_XBF    = OFF_WT_OUT + (size_t)DMODEL*DIN*2;           // bf16 [2048][1024]
constexpr size_t OFF_XR     = OFF_XBF    + (size_t)ROWS*DMODEL*2;          // f32  [2048][4096]
constexpr size_t OFF_XCF    = OFF_XR     + (size_t)ROWS*(2*DIN)*4;         // f32  [2048][2048]
constexpr size_t OFF_XCBF   = OFF_XCF    + (size_t)ROWS*DIN*4;             // bf16 [2048][2048]
constexpr size_t OFF_XDBL   = OFF_XCBF   + (size_t)ROWS*DIN*2;             // f32  [2048][96]
constexpr size_t OFF_DINBF  = OFF_XDBL   + (size_t)ROWS*XDBLW*4;           // bf16 [2048][64]
constexpr size_t OFF_DELTA  = OFF_DINBF  + (size_t)ROWS*DTRANK*2;          // f32  [2048][2048]
constexpr size_t OFF_YGBF   = OFF_DELTA  + (size_t)ROWS*DIN*4;             // bf16 [2048][2048]
constexpr size_t WS_NEEDED  = OFF_YGBF   + (size_t)ROWS*DIN*2;             // ~98 MB

// ---------------- helpers ----------------
__device__ __forceinline__ bf16 f2bf(float x) {
  union { unsigned short s; bf16 b; } cv;
  unsigned int u = __float_as_uint(x);
  unsigned int lsb = (u >> 16) & 1u;
  cv.s = (unsigned short)((u + 0x7FFFu + lsb) >> 16);   // round-to-nearest-even
  return cv.b;
}
__device__ __forceinline__ float silu(float v) { return v / (1.0f + __expf(-v)); }

// ---------------- kernels ----------------

// Wt[n][k] = (bf16) W[k][n]
__global__ void transpose_bf16_kernel(const float* __restrict__ W, bf16* __restrict__ Wt,
                                      int K, int N) {
  int idx = blockIdx.x * blockDim.x + threadIdx.x;
  if (idx >= K * N) return;
  int k = idx / N, n = idx - k * N;
  Wt[(size_t)n * K + k] = f2bf(W[idx]);
}

__global__ void cvt_bf16_kernel(const float* __restrict__ in, bf16* __restrict__ out, int n) {
  int idx = blockIdx.x * blockDim.x + threadIdx.x;
  if (idx < n) out[idx] = f2bf(in[idx]);
}

// C[M][N] (f32, ldc) = A[M][K] (bf16, lda) * Bt[N][K]^T (bf16, ldb)
// Wave tile 32(M) x 64(N): 8 WMMAs per K-step from 2 A + 4 B fragments.
// Block = 4 waves (2x2) covering 64(M) x 128(N).
// Out-of-range N columns are CLAMPED at load (garbage stays in their own lanes,
// WMMA never mixes columns across lanes) and masked at store -> no divergence
// in the K loop.
// epilogue: 0 = none, 1 = softplus(acc + bias[n])
__global__ __launch_bounds__(128) void gemm_bf16_kernel(
    const bf16* __restrict__ A, int lda,
    const bf16* __restrict__ Bt, int ldb,
    float* __restrict__ C, int ldc,
    int M, int N, int K,
    const float* __restrict__ bias, int epilogue)
{
  const int lane = threadIdx.x & 31;
  const int wave = threadIdx.x >> 5;
  const int r    = lane & 15;
  const int half = lane >> 4;
  const int m0 = blockIdx.y * 64  + (wave >> 1) * 32;
  const int n0 = blockIdx.x * 128 + (wave & 1) * 64;

  // A fragment pointers (16x32 tile): lane r rows m0+r / m0+16+r,
  // K bytes: lanes 0-15 at +0/+32, lanes 16-31 at +16/+48 within a 32-K step.
  const bf16* pa0 = A + (size_t)(m0 + r) * lda + half * 8;
  const bf16* pa1 = pa0 + (size_t)16 * lda;

  // B fragment pointers (32x16 tiles): lane = column, K: lanes 0-15 at +0/+16, lanes 16-31 at +8/+24 (same as A).
  const bf16* pb[4];
  #pragma unroll
  for (int j = 0; j < 4; ++j) {
    int n = n0 + j * 16 + r;
    if (n > N - 1) n = N - 1;                 // clamp: masked at store
    pb[j] = Bt + (size_t)n * ldb + half * 8;
  }

  v8f acc[2][4];
  #pragma unroll
  for (int im = 0; im < 2; ++im)
    #pragma unroll
    for (int jn = 0; jn < 4; ++jn) {
      v8f z = {0,0,0,0,0,0,0,0};
      acc[im][jn] = z;
    }

  for (int k0 = 0; k0 < K; k0 += 32) {
    FragAB a0, a1, b[4];
    a0.u[0] = *(const u32x4*)(pa0 + k0);
    a0.u[1] = *(const u32x4*)(pa0 + k0 + 16);
    a1.u[0] = *(const u32x4*)(pa1 + k0);
    a1.u[1] = *(const u32x4*)(pa1 + k0 + 16);
    #pragma unroll
    for (int j = 0; j < 4; ++j) {
      b[j].u[0] = *(const u32x4*)(pb[j] + k0);
      b[j].u[1] = *(const u32x4*)(pb[j] + k0 + 16);
    }
    #pragma unroll
    for (int j = 0; j < 4; ++j) {
      acc[0][j] = __builtin_amdgcn_wmma_f32_16x16x32_bf16(false, a0.bf, false, b[j].bf, (short)0, acc[0][j], false, false);
      acc[1][j] = __builtin_amdgcn_wmma_f32_16x16x32_bf16(false, a1.bf, false, b[j].bf, (short)0, acc[1][j], false, false);
    }
  }

  // store: VGPR i of accumulator -> row (sub-tile base + half*8 + i), column lane r
  #pragma unroll
  for (int im = 0; im < 2; ++im) {
    #pragma unroll
    for (int jn = 0; jn < 4; ++jn) {
      int mb = m0 + im * 16 + half * 8;
      int nb = n0 + jn * 16 + r;
      if (nb >= N) continue;
      float bval = (epilogue == 1) ? bias[nb] : 0.0f;
      #pragma unroll
      for (int i = 0; i < 8; ++i) {
        int m = mb + i;
        if (m >= M) continue;
        float v = acc[im][jn][i];
        if (epilogue == 1) {
          v += bval;
          v = (v > 20.0f) ? v : log1pf(__expf(v));  // softplus
        }
        C[(size_t)m * ldc + nb] = v;
      }
    }
  }
}

// depthwise causal conv (width 4) + bias + SiLU over xr[:, 0:DIN]; writes f32 + bf16
__global__ void conv_silu_kernel(const float* __restrict__ xr,
                                 const float* __restrict__ conv_w,
                                 const float* __restrict__ conv_b,
                                 float* __restrict__ xc_f, bf16* __restrict__ xc_bf) {
  int idx = blockIdx.x * blockDim.x + threadIdx.x;
  if (idx >= ROWS * DIN) return;
  int row = idx / DIN, c = idx - row * DIN;
  int b = row / SEQ, l = row - b * SEQ;
  float acc = conv_b[c];
  #pragma unroll
  for (int j = 0; j < DCONV; ++j) {
    int ll = l - (DCONV - 1) + j;
    if (ll >= 0)
      acc += conv_w[j * DIN + c] * xr[(size_t)(b * SEQ + ll) * (2 * DIN) + c];
  }
  float s = silu(acc);
  xc_f[idx]  = s;
  xc_bf[idx] = f2bf(s);
}

// bf16 copy of x_dbl[:, 0:64] (delta projection input)
__global__ void extract_delta_in_kernel(const float* __restrict__ xdbl, bf16* __restrict__ din) {
  int idx = blockIdx.x * blockDim.x + threadIdx.x;
  if (idx >= ROWS * DTRANK) return;
  int r = idx >> 6, c = idx & 63;
  din[idx] = f2bf(xdbl[(size_t)r * XDBLW + c]);
}

// selective scan: one thread per (b, d); 16 states in registers; B/C staged via LDS.
// fuses  y = (scan + u*D) * silu(res)  and stores bf16 for the output GEMM.
__global__ __launch_bounds__(256) void scan_kernel(
    const float* __restrict__ delta,   // [ROWS][DIN]
    const float* __restrict__ xconv,   // [ROWS][DIN]  (u)
    const float* __restrict__ xdbl,    // [ROWS][96]  cols 64..79 = B, 80..95 = C
    const float* __restrict__ xr,      // [ROWS][4096] cols 2048.. = res
    const float* __restrict__ A_log,   // [DIN][16]
    const float* __restrict__ Dvec,    // [DIN]
    bf16* __restrict__ yg)             // [ROWS][DIN]
{
  constexpr int CHUNK = 8;
  __shared__ float sBC[CHUNK][2 * DSTATE];   // 8 x 32 floats

  const int d = blockIdx.x * blockDim.x + threadIdx.x;   // 0..2047
  const int b = blockIdx.y;
  const int tid = threadIdx.x;

  float a[DSTATE], st[DSTATE];
  #pragma unroll
  for (int n = 0; n < DSTATE; ++n) {
    a[n]  = -__expf(A_log[(size_t)d * DSTATE + n]);
    st[n] = 0.0f;
  }
  const float Dd = Dvec[d];

  for (int lc = 0; lc < SEQ; lc += CHUNK) {
    __syncthreads();
    {
      int i = tid >> 5, j = tid & 31;   // 256 threads -> 8 x 32 values
      sBC[i][j] = xdbl[(size_t)(b * SEQ + lc + i) * XDBLW + DTRANK + j];
    }
    __syncthreads();

    for (int i = 0; i < CHUNK; ++i) {
      const size_t row = (size_t)(b * SEQ + lc + i);
      const float dl = delta[row * DIN + d];
      const float u  = xconv[row * DIN + d];
      const float du = dl * u;
      float y = 0.0f;
      #pragma unroll
      for (int n = 0; n < DSTATE; ++n) {
        float dA = __expf(dl * a[n]);
        st[n] = dA * st[n] + du * sBC[i][n];
        y += st[n] * sBC[i][DSTATE + n];
      }
      y += u * Dd;
      float res = xr[row * (2 * DIN) + DIN + d];
      yg[row * DIN + d] = f2bf(y * silu(res));
    }
  }
}

// ---------------- launch ----------------
extern "C" void kernel_launch(void* const* d_in, const int* in_sizes, int n_in,
                              void* d_out, int out_size, void* d_ws, size_t ws_size,
                              hipStream_t stream) {
  const float* x      = (const float*)d_in[0];
  const float* W_in   = (const float*)d_in[1];
  const float* conv_w = (const float*)d_in[2];
  const float* conv_b = (const float*)d_in[3];
  const float* W_x    = (const float*)d_in[4];
  const float* W_dt   = (const float*)d_in[5];
  const float* b_dt   = (const float*)d_in[6];
  const float* A_log  = (const float*)d_in[7];
  const float* Dvec   = (const float*)d_in[8];
  const float* W_out  = (const float*)d_in[9];
  float* out = (float*)d_out;

  char* w = (char*)d_ws;
  bf16*  Wt_in  = (bf16*)(w + OFF_WT_IN);
  bf16*  Wt_x   = (bf16*)(w + OFF_WT_X);
  bf16*  Wt_dt  = (bf16*)(w + OFF_WT_DT);
  bf16*  Wt_out = (bf16*)(w + OFF_WT_OUT);
  bf16*  x_bf   = (bf16*)(w + OFF_XBF);
  float* xr     = (float*)(w + OFF_XR);
  float* xc_f   = (float*)(w + OFF_XCF);
  bf16*  xc_bf  = (bf16*)(w + OFF_XCBF);
  float* xdbl   = (float*)(w + OFF_XDBL);
  bf16*  din_bf = (bf16*)(w + OFF_DINBF);
  float* delta  = (float*)(w + OFF_DELTA);
  bf16*  yg_bf  = (bf16*)(w + OFF_YGBF);

  const int TB = 256;
  auto blocks = [](long long n, int tb) { return (unsigned)((n + tb - 1) / tb); };

  // weight transposes + activation convert (bf16)
  transpose_bf16_kernel<<<blocks((long long)DMODEL * 2 * DIN, TB), TB, 0, stream>>>(W_in,  Wt_in,  DMODEL, 2 * DIN);
  transpose_bf16_kernel<<<blocks((long long)DIN * XDBLW,      TB), TB, 0, stream>>>(W_x,   Wt_x,   DIN,    XDBLW);
  transpose_bf16_kernel<<<blocks((long long)DTRANK * DIN,     TB), TB, 0, stream>>>(W_dt,  Wt_dt,  DTRANK, DIN);
  transpose_bf16_kernel<<<blocks((long long)DIN * DMODEL,     TB), TB, 0, stream>>>(W_out, Wt_out, DIN,    DMODEL);
  cvt_bf16_kernel<<<blocks((long long)ROWS * DMODEL, TB), TB, 0, stream>>>(x, x_bf, ROWS * DMODEL);

  // GEMM1: xr[2048][4096] = x_bf @ W_in
  {
    dim3 g((2 * DIN) / 128, ROWS / 64);
    gemm_bf16_kernel<<<g, 128, 0, stream>>>(x_bf, DMODEL, Wt_in, DMODEL,
                                            xr, 2 * DIN, ROWS, 2 * DIN, DMODEL, nullptr, 0);
  }

  // depthwise causal conv + SiLU
  conv_silu_kernel<<<blocks((long long)ROWS * DIN, TB), TB, 0, stream>>>(xr, conv_w, conv_b, xc_f, xc_bf);

  // GEMM2: x_dbl[2048][96] = x_conv @ W_x   (N=96 -> one 128-wide block column, clamp+mask)
  {
    dim3 g((XDBLW + 127) / 128, ROWS / 64);
    gemm_bf16_kernel<<<g, 128, 0, stream>>>(xc_bf, DIN, Wt_x, DIN,
                                            xdbl, XDBLW, ROWS, XDBLW, DIN, nullptr, 0);
  }

  extract_delta_in_kernel<<<blocks((long long)ROWS * DTRANK, TB), TB, 0, stream>>>(xdbl, din_bf);

  // GEMM3: delta = softplus(x_dbl[:, :64] @ W_dt + b_dt)
  {
    dim3 g(DIN / 128, ROWS / 64);
    gemm_bf16_kernel<<<g, 128, 0, stream>>>(din_bf, DTRANK, Wt_dt, DTRANK,
                                            delta, DIN, ROWS, DIN, DTRANK, b_dt, 1);
  }

  // selective scan (fused + u*D and * silu(res)), bf16 output
  {
    dim3 g(DIN / 256, BATCH);
    scan_kernel<<<g, 256, 0, stream>>>(delta, xc_f, xdbl, xr, A_log, Dvec, yg_bf);
  }

  // GEMM4: out[2048][1024] = yg @ W_out
  {
    dim3 g(DMODEL / 128, ROWS / 64);
    gemm_bf16_kernel<<<g, 128, 0, stream>>>(yg_bf, DIN, Wt_out, DIN,
                                            out, DMODEL, ROWS, DMODEL, DIN, nullptr, 0);
  }
}
